// GCN_23450521436961
// MI455X (gfx1250) — matrix operand order repaired and checked by reading, not compiled, not run on hardware
//
#include <hip/hip_runtime.h>
#include <hip/hip_bf16.h>
#include <stdint.h>

// ---------------- problem constants (match reference) ----------------
#define N_NODES   50000
#define N_EDGES   800000
#define D_INF     256
#define D_HIDF    256
#define D_OUTF    16
#define N_GRAPHS  128

// ---------------- WMMA vector types (gfx1250, wave32) ----------------
typedef __attribute__((ext_vector_type(16))) __bf16 v16bf;
typedef __attribute__((ext_vector_type(8)))  float  v8f;

union BFrag { v16bf v; uint4 q[2]; };

// workspace layout in float units
#define DEG_OFF   0ll
#define AGG_OFF   50048ll
#define SUMS_OFF  (AGG_OFF + 12800000ll)
#define CNTS_OFF  (SUMS_OFF + 32768ll)
#define H_OFF     (CNTS_OFF + 128ll)
#define XB_OFF    (H_OFF + 12800000ll)            // bf16 X: 12.8M bf16 = 6.4M floats
#define WT_OFF    (XB_OFF + 6400000ll)            // bf16 W^T: 65536 bf16 = 32768 floats
#define NZERO     (12800000ll + 32768ll + 128ll)  // AGG|sums|cnts contiguous

__device__ __forceinline__ uint32_t hash_u32(uint32_t x) {
    x ^= x >> 16; x *= 0x7feb352dU;
    x ^= x >> 15; x *= 0x846ca68bU;
    x ^= x >> 16;
    return x;
}

// ---------------- init: deg = 1.0 (self-loop weight) ----------------
__global__ void k_init_deg(float* __restrict__ deg) {
    int i = blockIdx.x * 256 + threadIdx.x;
    if (i < N_NODES) deg[i] = 1.0f;
}

// ---------------- zero AGG | sums | cnts ----------------
__global__ void k_zero(float* __restrict__ z, long long n) {
    long long i = (long long)blockIdx.x * 256 + threadIdx.x;
    if (i < n) z[i] = 0.0f;
}

// ---------------- deg[col] += ew ----------------
__global__ void k_degree(const long long* __restrict__ col,
                         const float* __restrict__ ew,
                         float* __restrict__ deg) {
    int e = blockIdx.x * 256 + threadIdx.x;
    if (e < N_EDGES) atomicAdd(&deg[(int)col[e]], ew[e]);
}

// ---------------- deg -> rsqrt(deg) in place ----------------
__global__ void k_dinv(float* __restrict__ deg) {
    int i = blockIdx.x * 256 + threadIdx.x;
    if (i < N_NODES) {
        float d = deg[i];
        deg[i] = (d > 0.0f) ? __frsqrt_rn(d) : 0.0f;
    }
}

// ---------------- X (f32) -> Xb (bf16), row-major ----------------
// each thread converts 8 consecutive floats -> 8 bf16 (one 16B store)
__global__ void __launch_bounds__(256)
k_cvt_x(const float* __restrict__ X, __bf16* __restrict__ Xb) {
    const long long t = (long long)blockIdx.x * 256 + threadIdx.x;
    const long long base = t * 8;
    if (base >= (long long)N_NODES * D_INF) return;
    const float4* src = (const float4*)(X + base);
    float4 a = src[0], b = src[1];
    __bf16 o[8] = {(__bf16)a.x, (__bf16)a.y, (__bf16)a.z, (__bf16)a.w,
                   (__bf16)b.x, (__bf16)b.y, (__bf16)b.z, (__bf16)b.w};
    *(uint4*)(Xb + base) = *(const uint4*)o;
}

// ---------------- W (f32, [k][n]) -> Wt (bf16, [n][k]) ----------------
__global__ void __launch_bounds__(256)
k_cvt_wt(const float* __restrict__ W, __bf16* __restrict__ Wt) {
    const int t = blockIdx.x * 256 + threadIdx.x;   // 32768 threads
    if (t >= D_HIDF * (D_INF / 2)) return;
    const int n  = t >> 7;          // output row (0..255)
    const int kp = t & 127;         // k pair (0..127)
    __bf16 o[2] = {(__bf16)W[(2 * kp) * D_HIDF + n],
                   (__bf16)W[(2 * kp + 1) * D_HIDF + n]};
    *(uint32_t*)(Wt + n * D_INF + 2 * kp) = *(const uint32_t*)o;
}

// ---------------- H = X @ W1 via v_wmma_f32_16x16x32_bf16 ----------------
// Block: 128 threads = 4 waves. Each wave computes one 16x256 output
// row-block (16 N-tiles, v8f acc[16]).  Whole Wt (bf16) is staged in LDS
// once per block (row pitch 264 bf16 = 132 dwords -> bank stride 4).
// Per k-step: 2 global b128 A-loads, 32 ds_load_b128 B-loads, 16 WMMAs.
#define WT_PITCH_DW 132
__global__ void __launch_bounds__(128)
k_gemm_xw(const __bf16* __restrict__ Xb,     // [N_NODES, 256] bf16
          const __bf16* __restrict__ Wt,     // [256, 256] bf16 (W^T)
          float* __restrict__ H) {           // [N_NODES, 256] f32
    __shared__ __align__(16) uint32_t ldsW[256 * WT_PITCH_DW];  // 132 KB

    const int tid  = threadIdx.x;
    // cooperative copy Wt -> LDS as uint4 (8192 uint4 total, 64 per thread)
    {
        const uint4* src = (const uint4*)Wt;      // 8192 uint4, row = 32 uint4
        #pragma unroll 1
        for (int idx = tid; idx < 8192; idx += 128) {
            const int row  = idx >> 5;            // 0..255
            const int col4 = idx & 31;            // 0..31
            *(uint4*)&ldsW[row * WT_PITCH_DW + col4 * 4] = src[idx];
        }
    }
    __syncthreads();

    const int wave = tid >> 5;
    const int tm   = blockIdx.x * 4 + wave;       // M tile 0..3124
    if (tm >= N_NODES / 16) return;
    const int lane = tid & 31;
    const int half = lane >> 4;
    const int r    = lane & 15;                   // A row / C column index

    v8f acc[16];
    #pragma unroll
    for (int t = 0; t < 16; ++t) acc[t] = (v8f){};

    const __bf16* xrow = Xb + (long long)(tm * 16 + r) * D_INF;

    #pragma unroll 1
    for (int k0 = 0; k0 < D_INF; k0 += 32) {
        // A fragment: K = [16h .. 16h+15]
        BFrag a;
        a.q[0] = *(const uint4*)(xrow + k0 + 16 * half);
        a.q[1] = *(const uint4*)(xrow + k0 + 16 * half + 8);

        #pragma unroll
        for (int tn = 0; tn < 16; ++tn) {
            // B fragment from LDS: row (tn*16 + c), K = [k0+16h .. +15]
            BFrag b;
            const uint32_t off = (uint32_t)(tn * 16 + r) * WT_PITCH_DW
                               + (uint32_t)(k0 / 2) + 8u * half;
            const uint4* p = (const uint4*)&ldsW[off];
            b.q[0] = p[0];
            b.q[1] = p[1];
            acc[tn] = __builtin_amdgcn_wmma_f32_16x16x32_bf16(
                false, a.v, false, b.v, (short)0, acc[tn], false, false);
        }
    }

    // store: acc[tn][i] -> H[(tm*16 + i + 8h)][tn*16 + r]
    #pragma unroll
    for (int tn = 0; tn < 16; ++tn) {
        float* hb = H + (long long)(tm * 16 + 8 * half) * D_HIDF + tn * 16 + r;
        #pragma unroll
        for (int i = 0; i < 8; ++i) hb[i * D_HIDF] = acc[tn][i];
    }
}

// ---------------- AGG[col] += norm * H[row]  (edges + self loops) ----------------
__global__ void __launch_bounds__(256)
k_prop(const long long* __restrict__ row,
       const long long* __restrict__ col,
       const float* __restrict__ ew,
       const float* __restrict__ dinv,
       const float* __restrict__ H,
       float* __restrict__ AGG) {
    const int item = blockIdx.x * 8 + (threadIdx.x >> 5);
    const int lane = threadIdx.x & 31;
    int src, dst; float w;
    if (item < N_EDGES) {
        src = (int)row[item];
        dst = (int)col[item];
        w   = ew[item];
    } else {
        int n = item - N_EDGES;
        if (n >= N_NODES) return;
        src = dst = n;
        w = 1.0f;
    }
    const float norm = dinv[src] * w * dinv[dst];
    const float4* hs = (const float4*)(H + (long long)src * D_HIDF);
    float* out = AGG + (long long)dst * D_HIDF;
    #pragma unroll
    for (int j = 0; j < 2; ++j) {
        const int q = lane + j * 32;
        float4 v = hs[q];
        const int base = q * 4;
        atomicAdd(out + base + 0, norm * v.x);
        atomicAdd(out + base + 1, norm * v.y);
        atomicAdd(out + base + 2, norm * v.z);
        atomicAdd(out + base + 3, norm * v.w);
    }
}

// ---------------- relu + dropout(0.5) + segment-sum pool ----------------
__global__ void __launch_bounds__(256)
k_pool(const float* __restrict__ AGG,
       const float* __restrict__ b1,
       const long long* __restrict__ gidx,
       float* __restrict__ sums,
       float* __restrict__ cnts) {
    const int n = blockIdx.x;
    const int d = threadIdx.x;
    const int g = (int)gidx[n];
    float v = AGG[(long long)n * D_HIDF + d] + b1[d];
    v = v > 0.0f ? v : 0.0f;
    uint32_t h = hash_u32((uint32_t)(n * D_HIDF + d) ^ 0xC0FFEE42u);
    v = (h & 1u) ? (v * 2.0f) : 0.0f;
    atomicAdd(&sums[(long long)g * D_HIDF + d], v);
    if (d == 0) atomicAdd(&cnts[g], 1.0f);
}

// ---------------- out = (sums / max(cnt,1)) @ fcW + fcb ----------------
__global__ void __launch_bounds__(256)
k_final(const float* __restrict__ sums,
        const float* __restrict__ cnts,
        const float* __restrict__ fcW,
        const float* __restrict__ fcb,
        float* __restrict__ out) {
    const int t = blockIdx.x * 256 + threadIdx.x;
    if (t >= N_GRAPHS * D_OUTF) return;
    const int g = t / D_OUTF;
    const int o = t % D_OUTF;
    float c = cnts[g];
    c = c > 1.0f ? c : 1.0f;
    const float inv = 1.0f / c;
    float acc = 0.0f;
    #pragma unroll 4
    for (int k = 0; k < D_HIDF; ++k) {
        acc += (sums[(long long)g * D_HIDF + k] * inv) * fcW[k * D_OUTF + o];
    }
    out[t] = acc + fcb[o];
}

// ---------------- host-side launcher ----------------
extern "C" void kernel_launch(void* const* d_in, const int* in_sizes, int n_in,
                              void* d_out, int out_size, void* d_ws, size_t ws_size,
                              hipStream_t stream) {
    const float*     x    = (const float*)d_in[0];
    const long long* eidx = (const long long*)d_in[1];
    const float*     ew   = (const float*)d_in[2];
    const long long* gidx = (const long long*)d_in[3];
    const float*     W1   = (const float*)d_in[4];
    const float*     b1   = (const float*)d_in[5];
    const float*     fcW  = (const float*)d_in[6];
    const float*     fcb  = (const float*)d_in[7];
    float* out = (float*)d_out;

    float*  ws   = (float*)d_ws;
    float*  deg  = ws + DEG_OFF;
    float*  AGG  = ws + AGG_OFF;
    float*  sums = ws + SUMS_OFF;
    float*  cnts = ws + CNTS_OFF;
    float*  H    = ws + H_OFF;
    __bf16* Xb   = (__bf16*)(ws + XB_OFF);
    __bf16* Wt   = (__bf16*)(ws + WT_OFF);

    const long long* erow = eidx;
    const long long* ecol = eidx + N_EDGES;

    // 1) deg = 1 (self loops), zero AGG|sums|cnts
    k_init_deg<<<(N_NODES + 255) / 256, 256, 0, stream>>>(deg);
    k_zero<<<(int)((NZERO + 255) / 256), 256, 0, stream>>>(AGG, NZERO);

    // 2) deg[col] += ew ; deg -> rsqrt(deg)
    k_degree<<<(N_EDGES + 255) / 256, 256, 0, stream>>>(ecol, ew, deg);
    k_dinv<<<(N_NODES + 255) / 256, 256, 0, stream>>>(deg);

    // 3) precision conversion: X->bf16, W1->bf16 transposed
    k_cvt_x<<<(N_NODES * D_INF / 8 + 255) / 256, 256, 0, stream>>>(x, Xb);
    k_cvt_wt<<<(D_HIDF * (D_INF / 2) + 255) / 256, 256, 0, stream>>>(W1, Wt);

    // 4) H = Xb @ W1 (bf16 WMMA, f32 accum); 4 waves/block, 782 blocks
    k_gemm_xw<<<(N_NODES / 16 + 3) / 4, 128, 0, stream>>>(Xb, Wt, H);

    // 5) propagation: 800k edges + 50k self loops, one wave each
    {
        const int items = N_EDGES + N_NODES;
        k_prop<<<(items + 7) / 8, 256, 0, stream>>>(erow, ecol, ew, deg, H, AGG);
    }

    // 6) relu + dropout + pooled segment sums / counts
    k_pool<<<N_NODES, 256, 0, stream>>>(AGG, b1, gidx, sums, cnts);

    // 7) final linear
    k_final<<<(N_GRAPHS * D_OUTF + 255) / 256, 256, 0, stream>>>(sums, cnts, fcW, fcb, out);
}
